// NNConvClassifier_47227460387297
// MI455X (gfx1250) — matrix operand order repaired and checked by reading, not compiled, not run on hardware
//
#include <hip/hip_runtime.h>
#include <hip/hip_bf16.h>

typedef __attribute__((ext_vector_type(16))) _Float16 v16h;
typedef __attribute__((ext_vector_type(8)))  _Float16 v8h;
typedef __attribute__((ext_vector_type(2)))  _Float16 h2;
typedef __attribute__((ext_vector_type(8)))  float    v8f;

#define NN   20000
#define EE   160000
#define HID  32
#define EHID 64
#define EFD  95
#define EFP  96     // padded edge-feat K
#define GG   64
#define WAVES_PER_BLK 8

// ---------------- WMMA fragment helpers (per CDNA5 ISA layout tables) -------

// A-matrix 16x32 f16: lanes 0-15 row M=lane, halves 0-7 = K 0..7, halves 8-15 = K 16..23
//                     lanes 16-31 row M=lane-16, K 8..15 and 24..31
__device__ inline v16h load_frag_A(const _Float16* tile, int strideH, int kbase, int lane) {
  int row = lane & 15;
  int ko  = (lane & 16) ? 8 : 0;
  const _Float16* p = tile + row * strideH + kbase + ko;
  v8h lo = *(const v8h*)p;          // K = ko .. ko+7
  v8h hi = *(const v8h*)(p + 16);   // K = ko+16 .. ko+23
  v16h r;
#pragma unroll
  for (int i = 0; i < 8; ++i) { r[i] = lo[i]; r[i + 8] = hi[i]; }
  return r;
}

// B-matrix 32x16 f16 stored transposed (N-major: tileT[n*strideH + k]):
// lane holds column N=lane%16; lanes 0-15 K 0..7,16..23, lanes 16-31 K 8..15,24..31 (matches A)
__device__ inline v16h load_frag_B(const _Float16* tileT, int strideH, int nbase, int kbase, int lane) {
  int n  = nbase + (lane & 15);
  int ko = (lane & 16) ? 8 : 0;
  const _Float16* p = tileT + n * strideH + kbase + ko;
  v8h lo = *(const v8h*)p;
  v8h hi = *(const v8h*)(p + 16);
  v16h r;
#pragma unroll
  for (int i = 0; i < 8; ++i) { r[i] = lo[i]; r[i + 8] = hi[i]; }
  return r;
}

__device__ inline v16h scale_frag(v16h a, _Float16 s) {
  h2 sb; sb[0] = s; sb[1] = s;
  v16h r;
  h2* ap = (h2*)&a; h2* rp = (h2*)&r;
#pragma unroll
  for (int i = 0; i < 8; ++i) rp[i] = ap[i] * sb;   // v_pk_mul_f16
  return r;
}

#define WMMA(a, b, c) __builtin_amdgcn_wmma_f32_16x16x32_f16(false, (a), false, (b), (short)0, (c), false, false)
#define LDS_FENCE() asm volatile("s_wait_dscnt 0x0" ::: "memory")

// ---------------- prep kernels ----------------------------------------------

__global__ void prep_bulk(const float* __restrict__ attr, const float* __restrict__ x,
                          _Float16* __restrict__ attrF, _Float16* __restrict__ x0f,
                          float* __restrict__ agg0, float* __restrict__ agg1,
                          int* __restrict__ pooled) {
  long long stride = (long long)gridDim.x * blockDim.x;
  long long tid = (long long)blockIdx.x * blockDim.x + threadIdx.x;
  for (long long i = tid; i < (long long)EE * EFP; i += stride) {
    int e = (int)(i / EFP), c = (int)(i % EFP);
    attrF[i] = (c < EFD) ? (_Float16)attr[(long long)e * EFD + c] : (_Float16)0.0f;
  }
  for (long long i = tid; i < (long long)NN * HID; i += stride) {
    x0f[i] = (_Float16)x[i];
    agg0[i] = 0.0f;
    agg1[i] = 0.0f;
  }
  for (long long i = tid; i < GG * HID; i += stride) pooled[i] = 0;  // bits of 0.0f
}

__global__ void prep_weights(const float* __restrict__ W1, const float* __restrict__ b1,
                             const float* __restrict__ W2, const float* __restrict__ b2,
                             const float* __restrict__ root, const float* __restrict__ cbias,
                             const float* __restrict__ bng, const float* __restrict__ bnb,
                             const float* __restrict__ rm, const float* __restrict__ rv,
                             _Float16* __restrict__ W1T, float* __restrict__ b1o,
                             _Float16* __restrict__ W2R, _Float16* __restrict__ rootT,
                             float* __restrict__ scale, float* __restrict__ shift) {
  // W1 (95x64) -> W1T[n*96 + k] (zero-padded K)
  for (int i = threadIdx.x; i < EHID * EFP; i += blockDim.x) {
    int n = i / EFP, k = i % EFP;
    W1T[i] = (k < EFD) ? (_Float16)W1[k * EHID + n] : (_Float16)0.0f;
  }
  for (int i = threadIdx.x; i < EHID; i += blockDim.x) b1o[i] = b1[i];
  // W2 (64 x 1024) with col = i*32+o  ->  W2R[kk][o][t] = W2[kk, t*32+o]; slice 64 = b2
  for (int i = threadIdx.x; i < 65 * 1024; i += blockDim.x) {
    int kk = i >> 10, o = (i >> 5) & 31, t = i & 31;
    W2R[i] = (kk < 64) ? (_Float16)W2[kk * 1024 + t * 32 + o] : (_Float16)b2[t * 32 + o];
  }
  // root (32x32, [i][o]) -> rootT[o*32 + i]
  for (int i = threadIdx.x; i < 1024; i += blockDim.x) {
    int o = i >> 5, ii = i & 31;
    rootT[i] = (_Float16)root[ii * 32 + o];
  }
  for (int o = threadIdx.x; o < HID; o += blockDim.x) {
    float s = bng[o] * rsqrtf(rv[o] + 1e-5f);
    scale[o] = s;
    shift[o] = (cbias[o] - rm[o]) * s + bnb[o];
  }
}

// ---------------- fused edge-pipeline conv kernel ---------------------------
// per 32-edge tile (one wave, two 16-row M-subtiles):
//   h  = relu(attr_tile(32x96) @ W1(96x64) + b1)
//   msg = sum_kk h[:,kk] * (xs_tile(32x32) @ W2slice_kk) (+ b2 slice)
//   scatter-add msg into agg[dst]
// B fragments per kk-slice are loaded once and shared by 4 independent WMMA chains.
__global__ __launch_bounds__(256) void conv_kernel(
    const _Float16* __restrict__ attrF,  // E x 96
    const int* __restrict__ srcIdx, const int* __restrict__ dstIdx,
    const _Float16* __restrict__ xF,     // N x 32
    const _Float16* __restrict__ W1T,    // 64 x 96  (N-major)
    const float* __restrict__ b1v,       // 64
    const _Float16* __restrict__ W2R,    // 65 x 32 x 32 ([kk][o][t])
    float* __restrict__ agg,             // N x 32
    int numTiles) {
  __shared__ _Float16 sW1T[EHID * EFP];            // 12 KB
  __shared__ _Float16 sW2R[65 * 1024];             // 130 KB
  __shared__ _Float16 sAttr[WAVES_PER_BLK][32 * EFP];
  __shared__ _Float16 sXs[WAVES_PER_BLK][32 * HID];
  __shared__ _Float16 sH[WAVES_PER_BLK][32 * EHID];

  for (int i = threadIdx.x; i < (EHID * EFP) / 8; i += blockDim.x)
    ((v8h*)sW1T)[i] = ((const v8h*)W1T)[i];
  for (int i = threadIdx.x; i < (65 * 1024) / 8; i += blockDim.x)
    ((v8h*)sW2R)[i] = ((const v8h*)W2R)[i];
  __syncthreads();

  const int lane = threadIdx.x & 31;
  const int wave = threadIdx.x >> 5;
  const int gw = blockIdx.x * WAVES_PER_BLK + wave;
  const int nW = gridDim.x * WAVES_PER_BLK;
  const int mbase = (lane & 16) ? 8 : 0;
  const int nlo = lane & 15;

  // hoist loop-invariant per-lane b1 values (n = nt*16 + lane%16)
  float bv[4];
#pragma unroll
  for (int nt = 0; nt < 4; ++nt) bv[nt] = b1v[nt * 16 + nlo];

  for (int tile = gw; tile < numTiles; tile += nW) {
    const int base = tile * 32;
    // stage attr tile: 32 rows x 96 halves = 384 chunks of 8 halves
    for (int t = lane; t < 32 * 12; t += 32) {
      int r = t / 12, c = t % 12;
      ((v8h*)sAttr[wave])[r * 12 + c] = *(const v8h*)(attrF + (long long)(base + r) * EFP + c * 8);
    }
    // gather source node rows: 32 rows x 32 halves = 128 chunks
    for (int t = lane; t < 32 * 4; t += 32) {
      int r = t >> 2, c = t & 3;
      int s = srcIdx[base + r];
      ((v8h*)sXs[wave])[r * 4 + c] = *(const v8h*)(xF + (long long)s * HID + c * 8);
    }
    LDS_FENCE();

    // ---- GEMM1: h = relu(attr @ W1 + b1), two M-subtiles share each B frag ----
    _Float16* hT = sH[wave];
    {
      v16h aA[3], aB[3];
#pragma unroll
      for (int kk = 0; kk < 3; ++kk) {
        aA[kk] = load_frag_A(sAttr[wave], EFP, kk * 32, lane);
        aB[kk] = load_frag_A(sAttr[wave] + 16 * EFP, EFP, kk * 32, lane);
      }
#pragma unroll
      for (int nt = 0; nt < 4; ++nt) {
        v8f accA = {}, accB = {};
#pragma unroll
        for (int kk = 0; kk < 3; ++kk) {
          v16h b = load_frag_B(sW1T, EFP, nt * 16, kk * 32, lane);
          accA = WMMA(aA[kk], b, accA);
          accB = WMMA(aB[kk], b, accB);
        }
        int n = nt * 16 + nlo;
#pragma unroll
        for (int r = 0; r < 8; ++r) {
          float vA = accA[r] + bv[nt];
          float vB = accB[r] + bv[nt];
          vA = vA > 0.0f ? vA : 0.0f;
          vB = vB > 0.0f ? vB : 0.0f;
          hT[(mbase + r) * EHID + n]        = (_Float16)vA;
          hT[(16 + mbase + r) * EHID + n]   = (_Float16)vB;
        }
      }
    }
    LDS_FENCE();

    // ---- GEMM2 (fused theta+einsum): 4 independent WMMA chains ----
    v16h xsA = load_frag_A(sXs[wave], HID, 0, lane);
    v16h xsB = load_frag_A(sXs[wave] + 16 * HID, HID, 0, lane);
    const int rowA = lane & 15;
    v8f m0A = {}, m1A = {}, m0B = {}, m1B = {};
    for (int kk = 0; kk < 64; ++kk) {
      const _Float16* slice = sW2R + kk * 1024;
      v16h b0 = load_frag_B(slice, 32, 0, 0, lane);
      v16h b1f = load_frag_B(slice, 32, 16, 0, lane);
      _Float16 hsA = hT[rowA * EHID + kk];
      _Float16 hsB = hT[(rowA + 16) * EHID + kk];
      v16h aA = scale_frag(xsA, hsA);
      v16h aB = scale_frag(xsB, hsB);
      m0A = WMMA(aA, b0, m0A);
      m1A = WMMA(aA, b1f, m1A);
      m0B = WMMA(aB, b0, m0B);
      m1B = WMMA(aB, b1f, m1B);
    }
    {  // bias slice (scale 1.0)
      const _Float16* slice = sW2R + 64 * 1024;
      v16h b0 = load_frag_B(slice, 32, 0, 0, lane);
      v16h b1f = load_frag_B(slice, 32, 16, 0, lane);
      m0A = WMMA(xsA, b0, m0A);
      m1A = WMMA(xsA, b1f, m1A);
      m0B = WMMA(xsB, b0, m0B);
      m1B = WMMA(xsB, b1f, m1B);
    }

    // ---- scatter-add into agg[dst] ----
#pragma unroll
    for (int r = 0; r < 8; ++r) {
      int mA = mbase + r;
      int dA = dstIdx[base + mA];
      float* pA = agg + (long long)dA * HID;
      __hip_atomic_fetch_add(pA + nlo,      m0A[r], __ATOMIC_RELAXED, __HIP_MEMORY_SCOPE_AGENT);
      __hip_atomic_fetch_add(pA + nlo + 16, m1A[r], __ATOMIC_RELAXED, __HIP_MEMORY_SCOPE_AGENT);
      int dB = dstIdx[base + 16 + mA];
      float* pB = agg + (long long)dB * HID;
      __hip_atomic_fetch_add(pB + nlo,      m0B[r], __ATOMIC_RELAXED, __HIP_MEMORY_SCOPE_AGENT);
      __hip_atomic_fetch_add(pB + nlo + 16, m1B[r], __ATOMIC_RELAXED, __HIP_MEMORY_SCOPE_AGENT);
    }
  }
}

// ---------------- node update: relu(bn(agg + x@root + bias)) ----------------
__global__ __launch_bounds__(256) void node_kernel(
    const float* __restrict__ agg, const _Float16* __restrict__ xF,
    const _Float16* __restrict__ rootT,  // 32x32 N-major
    const float* __restrict__ scale, const float* __restrict__ shift,
    _Float16* __restrict__ xOut, const int* __restrict__ batch,
    int* __restrict__ pooled, int doPool, int numTiles) {
  __shared__ _Float16 sRoot[1024];
  __shared__ _Float16 sX[WAVES_PER_BLK][16 * HID];
  for (int i = threadIdx.x; i < 128; i += blockDim.x)
    ((v8h*)sRoot)[i] = ((const v8h*)rootT)[i];
  __syncthreads();

  const int lane = threadIdx.x & 31;
  const int wave = threadIdx.x >> 5;
  const int gw = blockIdx.x * WAVES_PER_BLK + wave;
  const int nW = gridDim.x * WAVES_PER_BLK;
  const int mbase = (lane & 16) ? 8 : 0;
  const int o = lane & 15;

  // hoist per-lane BN constants
  const float sc0 = scale[o],      sh0 = shift[o];
  const float sc1 = scale[o + 16], sh1 = shift[o + 16];

  // B fragments are loop-invariant: load once
  v16h b0 = load_frag_B(sRoot, 32, 0, 0, lane);
  v16h b1 = load_frag_B(sRoot, 32, 16, 0, lane);

  for (int tile = gw; tile < numTiles; tile += nW) {
    const int base = tile * 16;
    for (int t = lane; t < 64; t += 32)
      ((v8h*)sX[wave])[t] = *(const v8h*)(xF + (long long)base * HID + t * 8);
    LDS_FENCE();

    v16h a = load_frag_A(sX[wave], HID, 0, lane);
    v8f c0 = {}, c1 = {};
    c0 = WMMA(a, b0, c0);
    c1 = WMMA(a, b1, c1);

#pragma unroll
    for (int r = 0; r < 8; ++r) {
      int node = base + mbase + r;
      float v0 = c0[r] + agg[(long long)node * HID + o];
      float v1 = c1[r] + agg[(long long)node * HID + o + 16];
      v0 = v0 * sc0 + sh0;
      v1 = v1 * sc1 + sh1;
      v0 = v0 > 0.0f ? v0 : 0.0f;
      v1 = v1 > 0.0f ? v1 : 0.0f;
      xOut[(long long)node * HID + o]      = (_Float16)v0;
      xOut[(long long)node * HID + o + 16] = (_Float16)v1;
      if (doPool) {
        int g = batch[node];
        // post-relu values are >= 0, so int-bit compare == float compare
        __hip_atomic_fetch_max(pooled + g * HID + o,      __float_as_int(v0),
                               __ATOMIC_RELAXED, __HIP_MEMORY_SCOPE_AGENT);
        __hip_atomic_fetch_max(pooled + g * HID + o + 16, __float_as_int(v1),
                               __ATOMIC_RELAXED, __HIP_MEMORY_SCOPE_AGENT);
      }
    }
  }
}

// ---------------- tiny MLP head ---------------------------------------------
__global__ void head_kernel(const int* __restrict__ pooled,
                            const float* __restrict__ lin1W, const float* __restrict__ lin1b,
                            const float* __restrict__ lin2W, const float* __restrict__ lin2b,
                            float* __restrict__ out) {
  int g = threadIdx.x;
  if (g >= GG) return;
  float p[HID];
#pragma unroll
  for (int i = 0; i < HID; ++i) p[i] = __int_as_float(pooled[g * HID + i]);
  float h[HID];
#pragma unroll
  for (int o = 0; o < HID; ++o) {
    float s = lin1b[o];
#pragma unroll
    for (int i = 0; i < HID; ++i) s += p[i] * lin1W[i * HID + o];
    h[o] = s > 0.0f ? s : 0.0f;
  }
#pragma unroll
  for (int o = 0; o < 2; ++o) {
    float s = lin2b[o];
#pragma unroll
    for (int j = 0; j < HID; ++j) s += h[j] * lin2W[j * 2 + o];
    out[g * 2 + o] = s;
  }
}

// ---------------- host launch ------------------------------------------------

extern "C" void kernel_launch(void* const* d_in, const int* in_sizes, int n_in,
                              void* d_out, int out_size, void* d_ws, size_t ws_size,
                              hipStream_t stream) {
  const float* x        = (const float*)d_in[0];
  const int*   esrc     = (const int*)d_in[1];
  const int*   edst     = (const int*)d_in[2];
  const float* eattr    = (const float*)d_in[3];
  const int*   batch    = (const int*)d_in[4];
  const float* cW1[2]   = {(const float*)d_in[5],  (const float*)d_in[15]};
  const float* cb1[2]   = {(const float*)d_in[6],  (const float*)d_in[16]};
  const float* cW2[2]   = {(const float*)d_in[7],  (const float*)d_in[17]};
  const float* cb2[2]   = {(const float*)d_in[8],  (const float*)d_in[18]};
  const float* croot[2] = {(const float*)d_in[9],  (const float*)d_in[19]};
  const float* cbias[2] = {(const float*)d_in[10], (const float*)d_in[20]};
  const float* bng[2]   = {(const float*)d_in[11], (const float*)d_in[21]};
  const float* bnb[2]   = {(const float*)d_in[12], (const float*)d_in[22]};
  const float* bnrm[2]  = {(const float*)d_in[13], (const float*)d_in[23]};
  const float* bnrv[2]  = {(const float*)d_in[14], (const float*)d_in[24]};
  const float* lin1W = (const float*)d_in[25];
  const float* lin1b = (const float*)d_in[26];
  const float* lin2W = (const float*)d_in[27];
  const float* lin2b = (const float*)d_in[28];

  // workspace carving (256B aligned)
  char* ws = (char*)d_ws;
  size_t off = 0;
  auto carve = [&](size_t bytes) -> char* {
    char* p = ws + off;
    off = (off + bytes + 255) & ~(size_t)255;
    return p;
  };
  _Float16* attrF = (_Float16*)carve((size_t)EE * EFP * 2);
  _Float16* x0f   = (_Float16*)carve((size_t)NN * HID * 2);
  _Float16* x1f   = (_Float16*)carve((size_t)NN * HID * 2);
  _Float16* x2f   = (_Float16*)carve((size_t)NN * HID * 2);
  float*    agg0  = (float*)carve((size_t)NN * HID * 4);
  float*    agg1  = (float*)carve((size_t)NN * HID * 4);
  int*      pooled = (int*)carve((size_t)GG * HID * 4);
  _Float16* W1T[2];  float* b1o[2];  _Float16* W2R[2];  _Float16* rootT[2];
  float* scl[2]; float* shf[2];
  for (int l = 0; l < 2; ++l) {
    W1T[l]   = (_Float16*)carve((size_t)EHID * EFP * 2);
    b1o[l]   = (float*)carve(EHID * 4);
    W2R[l]   = (_Float16*)carve((size_t)65 * 1024 * 2);
    rootT[l] = (_Float16*)carve(1024 * 2);
    scl[l]   = (float*)carve(HID * 4);
    shf[l]   = (float*)carve(HID * 4);
  }

  const int numTilesE = EE / 32;   // 5000 (32-edge tiles)
  const int numTilesN = NN / 16;   // 1250

  prep_bulk<<<4096, 256, 0, stream>>>(eattr, x, attrF, x0f, agg0, agg1, pooled);
  for (int l = 0; l < 2; ++l)
    prep_weights<<<1, 256, 0, stream>>>(cW1[l], cb1[l], cW2[l], cb2[l], croot[l],
                                        cbias[l], bng[l], bnb[l], bnrm[l], bnrv[l],
                                        W1T[l], b1o[l], W2R[l], rootT[l], scl[l], shf[l]);

  conv_kernel<<<250, 256, 0, stream>>>(attrF, esrc, edst, x0f, W1T[0], b1o[0], W2R[0],
                                       agg0, numTilesE);
  node_kernel<<<80, 256, 0, stream>>>(agg0, x0f, rootT[0], scl[0], shf[0], x1f, batch,
                                      pooled, /*doPool=*/0, numTilesN);
  conv_kernel<<<250, 256, 0, stream>>>(attrF, esrc, edst, x1f, W1T[1], b1o[1], W2R[1],
                                       agg1, numTilesE);
  node_kernel<<<80, 256, 0, stream>>>(agg1, x1f, rootT[1], scl[1], shf[1], x2f, batch,
                                      pooled, /*doPool=*/1, numTilesN);
  head_kernel<<<1, 64, 0, stream>>>(pooled, lin1W, lin1b, lin2W, lin2b, (float*)d_out);
  (void)n_in; (void)in_sizes; (void)out_size; (void)ws_size;
}
